// CapsuleFC_66494683676927
// MI455X (gfx1250) — matrix operand order repaired and checked by reading, not compiled, not run on hardware
//
#include <hip/hip_runtime.h>
#include <math.h>

// Problem constants (from reference): B=256, NIN=64, DIN=128, M=32, DOUT=128
#define B_     256
#define NIN_   64
#define DIN_   128
#define M_     32
#define DOUT_  128
#define COLS_  (M_ * DOUT_)          // 4096 fused (m,d) columns
#define BT_    16                    // batch rows per workgroup (one WMMA M-tile)
#define SCALE_ 0.08838834764831845f  // 1/sqrt(DOUT)

// LDS layout for caps_phase: x tile (bf16) + votes slab (f32) + logits + weights
#define SM_X   (BT_ * DIN_ * 2)      //   4096 B
#define SM_V   (BT_ * COLS_ * 4)     // 262144 B
#define SM_LG  (BT_ * M_ * 4)        //   2048 B
#define SM_W   (BT_ * M_ * 4)        //   2048 B
#define SMEM_BYTES (SM_X + SM_V + SM_LG + SM_W)   // 270336 B (< 320 KB WGP LDS)

typedef __attribute__((ext_vector_type(16))) __bf16 v16bf;
typedef __attribute__((ext_vector_type(8)))  float  v8f;

__device__ __forceinline__ unsigned short f32_to_bf16(float f) {
  union { float f; unsigned int u; } v; v.f = f;
  unsigned int x = v.u;
  unsigned int lsb = (x >> 16) & 1u;
  x += 0x7fffu + lsb;               // round-to-nearest-even
  return (unsigned short)(x >> 16);
}

// ---------------------------------------------------------------- conversions
__global__ void cvt_x_kernel(const float* __restrict__ x,
                             unsigned short* __restrict__ xb, int n) {
  int i = blockIdx.x * blockDim.x + threadIdx.x;
  if (i < n) xb[i] = f32_to_bf16(x[i]);
}

// W [NIN][DIN][COLS] f32  ->  Wt [NIN][COLS][DIN] bf16 (column-major per n, so
// WMMA B-fragments become contiguous b128 loads per lane).
__global__ __launch_bounds__(256)
void transpose_w_kernel(const float* __restrict__ W,
                        unsigned short* __restrict__ Wt) {
  __shared__ unsigned short t[32][65];              // +1 pad: no bank conflicts
  const int n   = blockIdx.x >> 8;                  // 256 tiles per n
  const int rem = blockIdx.x & 255;
  const int a0  = (rem >> 6) * 32;                  // 4 a-tiles of 32
  const int c0  = (rem & 63) * 64;                  // 64 col-tiles of 64
  const int tid = threadIdx.x;
  for (int i = tid; i < 32 * 64; i += 256) {        // coalesced f32 reads
    int la = i >> 6, lc = i & 63;
    t[la][lc] = f32_to_bf16(W[((size_t)n * DIN_ + a0 + la) * COLS_ + c0 + lc]);
  }
  __syncthreads();
  for (int i = tid; i < 32 * 64; i += 256) {        // coalesced bf16 writes
    int lc = i >> 5, la = i & 31;
    Wt[((size_t)n * COLS_ + c0 + lc) * DIN_ + a0 + la] = t[la][lc];
  }
}

__global__ void zero_f32_kernel(float* __restrict__ p, int n) {
  int i = blockIdx.x * blockDim.x + threadIdx.x;
  if (i < n) p[i] = 0.0f;
}

__global__ void copy_f32_kernel(const float* __restrict__ s,
                                float* __restrict__ d, int n) {
  int i = blockIdx.x * blockDim.x + threadIdx.x;
  if (i < n) d[i] = s[i];
}

// ------------------------------------------------------------------ main pass
// One workgroup = (16-batch tile, capsule n). 8 waves recompute the 16x4096
// votes slab with v_wmma_f32_16x16x32_bf16 into 256 KB of LDS, then fuse the
// routing epilogue (logits / softmax / ncv atomic reduction / emb stores).
// phase: 0 = init (uniform 1/M weights), 1 = routing iter, 2 = final iter
//        (also writes route_class_emb).
__global__ __launch_bounds__(256)
void caps_phase_kernel(const unsigned short* __restrict__ xb,   // [B][NIN][DIN] bf16
                       const unsigned short* __restrict__ Wt,   // [NIN][COLS][DIN] bf16
                       const float* __restrict__ ncv_in,        // [B][COLS]
                       float* __restrict__ ncv_out,             // [B][COLS] (atomic)
                       float* __restrict__ q_out,               // [B][NIN][M]
                       float* __restrict__ emb_out,             // [B][NIN][COLS]
                       const float* __restrict__ act,           // [B][NIN]
                       int phase) {
  extern __shared__ char smem[];
  unsigned short* Xs = (unsigned short*)smem;
  float* votes = (float*)(smem + SM_X);
  float* lg    = (float*)(smem + SM_X + SM_V);
  float* wrow  = lg + BT_ * M_;

  const int n    = blockIdx.x & (NIN_ - 1);
  const int b0   = (blockIdx.x >> 6) * BT_;
  const int tid  = threadIdx.x;
  const int lane = tid & 31;                       // wave32
  const int wave = tid >> 5;                       // 8 waves

  // Stage x tile -> LDS: 16 rows x 128 bf16 = 256 uint4; one b128 per thread.
  {
    const int r   = tid >> 4;                      // 0..15
    const int c16 = tid & 15;                      // 16B chunk within row
    const uint4* src = (const uint4*)(xb + ((size_t)(b0 + r) * NIN_ + n) * DIN_);
    ((uint4*)(Xs + r * DIN_))[c16] = src[c16];
  }
  __syncthreads();

  // A fragments for full K=128 (reused across all 32 column tiles of this wave).
  // A 16x32 layout: lane L -> row L&15; lane half selects
  // K 0..15 (lanes 0-15) vs 16..31 (lanes 16-31), matching B.
  const int arow = lane & 15;
  const int aofs = (lane >> 4) * 16;
  v16bf afrag[4];
#pragma unroll
  for (int kc = 0; kc < 4; ++kc) {
    const uint4* s = (const uint4*)(Xs + arow * DIN_ + kc * 32 + aofs);
    uint4* ap = reinterpret_cast<uint4*>(&afrag[kc]);
    ap[0] = s[0];        // K +0..7
    ap[1] = s[1];        // K +8..15
  }

  // GEMM: [16 x 128] x [128 x 4096] -> votes slab in LDS (f32).
  // B 32x16 fragment, contiguous k order as in A: lane L -> col L&15; lanes 0-15 hold K 0..15,
  // lanes 16-31 hold K 16..31 -> 32 contiguous bytes per lane per K-chunk.
  const unsigned short* Wn = Wt + (size_t)n * COLS_ * DIN_;
  const int bcol_lo = lane & 15;
  const int bofs    = (lane >> 4) * 16;
  for (int t = 0; t < 32; ++t) {
    const int c0 = (wave * 32 + t) * 16;
    const unsigned short* wcol = Wn + (size_t)(c0 + bcol_lo) * DIN_ + bofs;
    if (t + 1 < 32)
      __builtin_prefetch(wcol + 16 * DIN_, 0, 3);  // WGP-scope prefetch, next tile
    v8f acc = {};
#pragma unroll
    for (int kc = 0; kc < 4; ++kc) {
      const uint4* p = (const uint4*)(wcol + kc * 32);
      v16bf bfrag;
      uint4* bp = reinterpret_cast<uint4*>(&bfrag);
      bp[0] = p[0];      // K +0..7   (relative to this lane-half's base)
      bp[1] = p[1];      // K +8..15
      acc = __builtin_amdgcn_wmma_f32_16x16x32_bf16(
          /*neg_a=*/false, afrag[kc], /*neg_b=*/false, bfrag,
          /*c_mod=*/(short)0, acc, /*reuse_a=*/false, /*reuse_b=*/false);
    }
    // D layout: vgpr r -> row r + 8*(lane>=16), col = c0 + (lane&15)
    const int vrow0 = (lane >> 4) * 8;
    const int vcol  = c0 + bcol_lo;
#pragma unroll
    for (int r = 0; r < 8; ++r)
      votes[(vrow0 + r) * COLS_ + vcol] = acc[r];
  }
  __syncthreads();

  if (phase == 0) {
    // ncv0 += votes / M  (cross-n reduction via L2-side f32 atomics)
    const float invM = 1.0f / (float)M_;
    for (int i = tid; i < BT_ * COLS_; i += 256) {
      int r = i >> 12, c = i & (COLS_ - 1);
      unsafeAtomicAdd(&ncv_out[(size_t)(b0 + r) * COLS_ + c], votes[i] * invM);
    }
    return;
  }

  // logits[b,m] = SCALE * <votes[b, m*128 : m*128+128], ncv_in[b,m,:]>
  for (int i = tid; i < BT_ * M_; i += 256) {
    int r = i >> 5, m = i & (M_ - 1);
    const float4* vp = (const float4*)(votes + r * COLS_ + m * DOUT_);
    const float4* np = (const float4*)(ncv_in + (size_t)(b0 + r) * COLS_ + m * DOUT_);
    float s = 0.0f;
    for (int d = 0; d < DOUT_ / 4; ++d) {
      float4 a4 = vp[d], b4 = np[d];
      s += a4.x * b4.x + a4.y * b4.y + a4.z * b4.z + a4.w * b4.w;
    }
    lg[i] = s * SCALE_;
  }
  __syncthreads();

  // softmax over M=32 per batch row; fold in activation -> routing weights
  if (tid < BT_) {
    float mx = -3.402823466e38f;
    for (int m = 0; m < M_; ++m) mx = fmaxf(mx, lg[tid * M_ + m]);
    float sum = 0.0f;
    for (int m = 0; m < M_; ++m) {
      float e = __expf(lg[tid * M_ + m] - mx);
      wrow[tid * M_ + m] = e;
      sum += e;
    }
    const float inv = 1.0f / sum;
    const float a   = act[(size_t)(b0 + tid) * NIN_ + n];
    for (int m = 0; m < M_; ++m) {
      float qv = wrow[tid * M_ + m] * inv;
      q_out[((size_t)(b0 + tid) * NIN_ + n) * M_ + m] = qv;
      wrow[tid * M_ + m] = qv * a;
    }
  }
  __syncthreads();

  // ncv_next += w * votes ; final phase also streams route_class_emb
  for (int i = tid; i < BT_ * COLS_; i += 256) {
    int r = i >> 12, c = i & (COLS_ - 1);
    float v = votes[i] * wrow[r * M_ + (c >> 7)];
    unsafeAtomicAdd(&ncv_out[(size_t)(b0 + r) * COLS_ + c], v);
    if (phase == 2)
      emb_out[((size_t)(b0 + r) * NIN_ + n) * COLS_ + c] = v;
  }
}

// ------------------------------------------------------------------- launcher
extern "C" void kernel_launch(void* const* d_in, const int* in_sizes, int n_in,
                              void* d_out, int out_size, void* d_ws, size_t ws_size,
                              hipStream_t stream) {
  (void)in_sizes; (void)n_in; (void)out_size; (void)ws_size;
  const float* x   = (const float*)d_in[0];        // [B,NIN,DIN]
  const float* act = (const float*)d_in[1];        // [B,NIN]
  const float* W   = (const float*)d_in[2];        // [NIN,DIN,M,DOUT]
  // d_in[3] = num_iter (device scalar; fixed at 3 to keep the graph static)

  // workspace carve-up (~147 MB)
  char* ws = (char*)d_ws;
  unsigned short* xb = (unsigned short*)ws;                         //   4 MB
  unsigned short* Wt = (unsigned short*)(ws + 4194304);             // 134 MB
  float* ncvA = (float*)(ws + 4194304 + 134217728);                 //   4 MB
  float* ncvB = ncvA + (size_t)B_ * COLS_;                          //   4 MB

  float* out     = (float*)d_out;
  float* out_ncv = out;                                   // [B,M,DOUT]
  float* out_q   = out + (size_t)B_ * COLS_;              // [B,NIN,M]
  float* out_emb = out_q + (size_t)B_ * NIN_ * M_;        // [B,NIN,M,DOUT]

  (void)hipFuncSetAttribute((const void*)caps_phase_kernel,
                            hipFuncAttributeMaxDynamicSharedMemorySize,
                            SMEM_BYTES);

  const int NCV = B_ * COLS_;                             // 1,048,576
  const int NX  = B_ * NIN_ * DIN_;                       // 2,097,152
  const dim3 blk(256);

  // 1) precision conversion + W transpose (one-time; bf16 W stays L2-resident)
  cvt_x_kernel<<<(NX + 255) / 256, blk, 0, stream>>>(x, xb, NX);
  transpose_w_kernel<<<NIN_ * 256, blk, 0, stream>>>(W, Wt);

  const int PG = (B_ / BT_) * NIN_;                       // 1024 workgroups

  // 2) init pass: ncv0 = votes averaged over m-weighting (uniform q = 1/M)
  zero_f32_kernel<<<NCV / 256, blk, 0, stream>>>(ncvA, NCV);
  caps_phase_kernel<<<PG, blk, SMEM_BYTES, stream>>>(
      xb, Wt, ncvA, ncvA, out_q, nullptr, act, /*phase=*/0);

  // 3) routing iteration 0: A -> B
  zero_f32_kernel<<<NCV / 256, blk, 0, stream>>>(ncvB, NCV);
  caps_phase_kernel<<<PG, blk, SMEM_BYTES, stream>>>(
      xb, Wt, ncvA, ncvB, out_q, nullptr, act, /*phase=*/1);

  // 4) routing iteration 1: B -> A
  zero_f32_kernel<<<NCV / 256, blk, 0, stream>>>(ncvA, NCV);
  caps_phase_kernel<<<PG, blk, SMEM_BYTES, stream>>>(
      xb, Wt, ncvB, ncvA, out_q, nullptr, act, /*phase=*/1);

  // 5) routing iteration 2 (final): A -> B, also emits q + route_class_emb
  zero_f32_kernel<<<NCV / 256, blk, 0, stream>>>(ncvB, NCV);
  caps_phase_kernel<<<PG, blk, SMEM_BYTES, stream>>>(
      xb, Wt, ncvA, ncvB, out_q, out_emb, act, /*phase=*/2);

  // 6) final ncv -> output
  copy_f32_kernel<<<NCV / 256, blk, 0, stream>>>(ncvB, out_ncv, NCV);
}
